// MalleConv_24111946400300
// MI455X (gfx1250) — compile-verified, matrix operand order checked by reading, not run on hardware
//
#include <hip/hip_runtime.h>
#include <hip/hip_bf16.h>
#include <math.h>

typedef __attribute__((ext_vector_type(16))) _Float16 v16h;
typedef __attribute__((ext_vector_type(8)))  _Float16 v8h;
typedef __attribute__((ext_vector_type(8)))  float    v8f;

union ABfrag { v16h v; v8h h[2]; };

// ---------------------------------------------------------------------------
// avgpool 4x4 : (4,32,512,512) -> (4,32,128,128)
// ---------------------------------------------------------------------------
__global__ __launch_bounds__(256) void avgpool4_k(const float* __restrict__ in,
                                                  float* __restrict__ out) {
    int idx = blockIdx.x * 256 + threadIdx.x;            // 4*32*128*128
    if (idx >= 4 * 32 * 128 * 128) return;
    int ow = idx & 127;
    int oh = (idx >> 7) & 127;
    int bc = idx >> 14;
    const float* p = in + ((size_t)bc * 512 + oh * 4) * 512 + ow * 4;
    float s = 0.f;
#pragma unroll
    for (int r = 0; r < 4; ++r) {
        float4 v = *(const float4*)(p + r * 512);
        s += v.x + v.y + v.z + v.w;
    }
    out[idx] = s * (1.f / 16.f);
}

// ---------------------------------------------------------------------------
// avgpool 2x2 : (4,32,128,128) -> (4,32,64,64)
// ---------------------------------------------------------------------------
__global__ __launch_bounds__(256) void avgpool2_k(const float* __restrict__ in,
                                                  float* __restrict__ out) {
    int idx = blockIdx.x * 256 + threadIdx.x;            // 4*32*64*64
    if (idx >= 4 * 32 * 64 * 64) return;
    int ow = idx & 63;
    int oh = (idx >> 6) & 63;
    int bc = idx >> 12;
    const float* p = in + ((size_t)bc * 128 + oh * 2) * 128 + ow * 2;
    float2 v0 = *(const float2*)p;
    float2 v1 = *(const float2*)(p + 128);
    out[idx] = (v0.x + v0.y + v1.x + v1.y) * 0.25f;
}

// ---------------------------------------------------------------------------
// conv3x3, C=32 -> 32 via WMMA (9 taps x K=32).
//  A = weights (M = out channel, 2 M-tiles), B = input pixels (N = 16 px).
//  mode 0: leaky_relu(0.01) ; mode 1: add residual.
//  tile: 16x16 pixels per workgroup, 256 threads (8 waves), halo 18x18.
// ---------------------------------------------------------------------------
__global__ __launch_bounds__(256) void conv3x3_wmma(
    const float* __restrict__ src, const float* __restrict__ resid,
    float* __restrict__ dst, const float* __restrict__ w,
    const float* __restrict__ bias, int H, int W, int mode) {
    __shared__ __align__(16) _Float16 s_in[18 * 18 * 32];   // [r][col][c]
    __shared__ __align__(16) _Float16 s_w[9 * 32 * 32];     // [tap][o][c]
    __shared__ float s_b[32];

    const int b = blockIdx.y;
    const int tilesX = W >> 4;
    const int tx = blockIdx.x % tilesX;
    const int ty = blockIdx.x / tilesX;
    const int h0 = ty * 16, w0 = tx * 16;
    const int t = threadIdx.x;

    // stage weights: [tap][o][c]
    for (int i = t; i < 9 * 32 * 32; i += 256) {
        int tap = i >> 10;
        int o = (i >> 5) & 31;
        int c = i & 31;
        s_w[i] = (_Float16)w[(o * 32 + c) * 9 + tap];
    }
    if (t < 32) s_b[t] = bias[t];

    // stage input tile with zero halo (coalesced over columns)
    for (int i = t; i < 32 * 18 * 18; i += 256) {
        int c = i / (18 * 18);
        int rem = i - c * (18 * 18);
        int r = rem / 18, col = rem % 18;
        int gh = h0 + r - 1, gw = w0 + col - 1;
        float v = 0.f;
        if (gh >= 0 && gh < H && gw >= 0 && gw < W)
            v = src[((size_t)(b * 32 + c) * H + gh) * W + gw];
        s_in[(r * 18 + col) * 32 + c] = (_Float16)v;
    }
    __syncthreads();

    const int wave = t >> 5;
    const int lane = t & 31;
    const int lhalf = lane >> 4;
    const int lmod = lane & 15;
    const int koff = lhalf ? 8 : 0;

    for (int nt = wave * 2; nt < wave * 2 + 2; ++nt) {   // N-tile = tile row
        v8f acc0, acc1;
#pragma unroll
        for (int r = 0; r < 8; ++r) {
            acc0[r] = s_b[r + lhalf * 8];
            acc1[r] = s_b[16 + r + lhalf * 8];
        }
#pragma unroll
        for (int tap = 0; tap < 9; ++tap) {
            int di = tap / 3, dj = tap % 3;
            ABfrag bf, af;
            const _Float16* bp = &s_in[((nt + di) * 18 + (lmod + dj)) * 32 + koff];
            bf.h[0] = *(const v8h*)bp;
            bf.h[1] = *(const v8h*)(bp + 16);
            const _Float16* ap = &s_w[(tap * 32 + lmod) * 32 + koff];
            af.h[0] = *(const v8h*)ap;
            af.h[1] = *(const v8h*)(ap + 16);
            acc0 = __builtin_amdgcn_wmma_f32_16x16x32_f16(
                false, af.v, false, bf.v, (short)0, acc0, false, false);
            const _Float16* ap1 = &s_w[(tap * 32 + 16 + lmod) * 32 + koff];
            af.h[0] = *(const v8h*)ap1;
            af.h[1] = *(const v8h*)(ap1 + 16);
            acc1 = __builtin_amdgcn_wmma_f32_16x16x32_f16(
                false, af.v, false, bf.v, (short)0, acc1, false, false);
        }
        // D: lane = pixel column, vgpr r = out channel (+8 for lanes>=16)
        int gh = h0 + nt;
        int gw = w0 + lmod;
#pragma unroll
        for (int r = 0; r < 8; ++r) {
            int ch0 = r + lhalf * 8;
            int ch1 = 16 + ch0;
            size_t i0 = ((size_t)(b * 32 + ch0) * H + gh) * W + gw;
            size_t i1 = ((size_t)(b * 32 + ch1) * H + gh) * W + gw;
            float v0 = acc0[r], v1 = acc1[r];
            if (mode == 0) {
                v0 = v0 > 0.f ? v0 : 0.01f * v0;
                v1 = v1 > 0.f ? v1 : 0.01f * v1;
            } else {
                v0 += resid[i0];
                v1 += resid[i1];
            }
            dst[i0] = v0;
            dst[i1] = v1;
        }
    }
}

// ---------------------------------------------------------------------------
// expand: conv1x1 32 -> 320 + tanh, stored as [b][c][hb*64+wb][kt] (kt contig)
// ---------------------------------------------------------------------------
__global__ __launch_bounds__(256) void expand_tanh_k(
    const float* __restrict__ feat, const float* __restrict__ ew,
    const float* __restrict__ eb, float* __restrict__ wmap) {
    int idx = blockIdx.x * 256 + threadIdx.x;            // 4*320*4096
    if (idx >= 4 * 320 * 4096) return;
    int p = idx & 4095;
    int oc = (idx >> 12) % 320;
    int b = idx / (320 * 4096);
    float acc = eb[oc];
#pragma unroll
    for (int c = 0; c < 32; ++c)
        acc += ew[oc * 32 + c] * feat[(size_t)((b * 32 + c) << 12) + p];
    acc = tanhf(acc);
    int cc = oc / 10, kt = oc % 10;
    wmap[((size_t)((b * 32 + cc) << 12) + p) * 10 + kt] = acc;
}

// ---------------------------------------------------------------------------
// fused flying_conv (per-8x8-block dynamic depthwise 3x3 + bias) + 1x1 conv.
// Tile = 8 rows x 32 cols (4 flying blocks), 256 threads, dynamic LDS.
//  x tile staged f32 via global_load_async_to_lds_b128 (ASYNCcnt path);
//  flying conv in fp32; y stored f16 -> WMMA B (N = pixels);
//  A = pw weights (M = out channel); coalesced D stores.
// LDS partition (bytes):
//   s_xf  [32][10][40] f32   @     0  (51200)   rows h0-1..h0+8, cols w0-4..w0+35
//   s_wm  [4][32][10]  f32   @ 51200  ( 5120)
//   s_y   [256][32]    f16   @ 56320  (16384)
//   s_pw  [32][32]     f16   @ 72704  ( 2048)
//   s_pb  [32]         f32   @ 74752  (  128)   total 74880
// ---------------------------------------------------------------------------
#define FLY_LDS_BYTES 74880

__global__ __launch_bounds__(256) void flying_pw_wmma(
    const float* __restrict__ x, const float* __restrict__ wmap,
    const float* __restrict__ pw_w, const float* __restrict__ pw_b,
    float* __restrict__ out) {
    extern __shared__ char smem[];
    float*    s_xf = (float*)smem;
    float*    s_wm = (float*)(smem + 51200);
    _Float16* s_y  = (_Float16*)(smem + 56320);
    _Float16* s_pw = (_Float16*)(smem + 72704);
    float*    s_pb = (float*)(smem + 74752);

    const int H = 512, W = 512;
    const int b = blockIdx.z;
    const int h0 = blockIdx.y * 8;
    const int w0 = blockIdx.x * 32;
    const int hb = blockIdx.y;
    const int wb0 = blockIdx.x * 4;
    const int t = threadIdx.x;

    // ---- stage x tile (f32) -------------------------------------------------
    const bool interior = (blockIdx.y >= 1) && (blockIdx.y <= 62) &&
                          (blockIdx.x >= 1) && (blockIdx.x <= 14);
    if (interior) {
        // async DMA: 32 ch x 10 rows x 10 chunks of 16B, all 16B aligned
        for (int i = t; i < 3200; i += 256) {
            int c = i / 100;
            int rem = i - c * 100;
            int r = rem / 10;
            int q = rem - r * 10;
            const float* gp =
                x + ((size_t)(b * 32 + c) * H + (h0 - 1 + r)) * W + (w0 - 4) + q * 4;
            unsigned int la =
                (unsigned int)(unsigned long long)(uintptr_t)&s_xf[(c * 10 + r) * 40 + q * 4];
            unsigned long long ga = (unsigned long long)(uintptr_t)gp;
            asm volatile("global_load_async_to_lds_b128 %0, %1, off"
                         :: "v"(la), "v"(ga) : "memory");
        }
    } else {
        // boundary: scalar loads with zero padding
        for (int i = t; i < 32 * 10 * 40; i += 256) {
            int c = i / 400;
            int rem = i - c * 400;
            int r = rem / 40;
            int q = rem - r * 40;
            int gh = h0 + r - 1, gw = w0 - 4 + q;
            float v = 0.f;
            if (gh >= 0 && gh < H && gw >= 0 && gw < W) {
                const float* gp = &x[((size_t)(b * 32 + c) * H + gh) * W + gw];
                v = *gp;
                __builtin_prefetch(gp + 32, 0, 1);
            }
            s_xf[i] = v;
        }
    }
    // ---- stage flying weights + pw (overlaps with async x DMA) -------------
    for (int i = t; i < 4 * 32 * 10; i += 256) {
        int j = i / 320;
        int c = (i / 10) & 31;
        int kt = i % 10;
        s_wm[(j * 32 + c) * 10 + kt] =
            wmap[(((size_t)(b * 32 + c) * 64 + hb) * 64 + (wb0 + j)) * 10 + kt];
    }
    for (int i = t; i < 1024; i += 256) s_pw[i] = (_Float16)pw_w[i];
    if (t < 32) s_pb[t] = pw_b[t];

    asm volatile("s_wait_asynccnt 0x0" ::: "memory");
    __syncthreads();

    // ---- flying conv in fp32: one pixel per thread --------------------------
    {
        int p = t;
        int r = p >> 5, col = p & 31;
        int j = col >> 3;
        for (int c = 0; c < 32; ++c) {
            const float* wmc = &s_wm[(j * 32 + c) * 10];
            const float* xc = &s_xf[(c * 10 + r) * 40 + 3 + col];  // (r+0, col-1)
            float acc = wmc[9];
            acc += xc[0]  * wmc[0] + xc[1]  * wmc[1] + xc[2]  * wmc[2];
            acc += xc[40] * wmc[3] + xc[41] * wmc[4] + xc[42] * wmc[5];
            acc += xc[80] * wmc[6] + xc[81] * wmc[7] + xc[82] * wmc[8];
            s_y[p * 32 + c] = (_Float16)acc;
        }
    }
    __syncthreads();

    // ---- 1x1 channel mix via WMMA ------------------------------------------
    const int wave = t >> 5;
    const int lane = t & 31;
    const int lhalf = lane >> 4;
    const int lmod = lane & 15;
    const int koff = lhalf ? 8 : 0;

    ABfrag a0, a1;
    {
        const _Float16* ap = &s_pw[lmod * 32 + koff];
        a0.h[0] = *(const v8h*)ap;
        a0.h[1] = *(const v8h*)(ap + 16);
        const _Float16* ap1 = &s_pw[(16 + lmod) * 32 + koff];
        a1.h[0] = *(const v8h*)ap1;
        a1.h[1] = *(const v8h*)(ap1 + 16);
    }

    for (int nt = wave * 2; nt < wave * 2 + 2; ++nt) {   // 16 N-tiles of 16 px
        v8f c0, c1;
#pragma unroll
        for (int r = 0; r < 8; ++r) {
            c0[r] = s_pb[r + lhalf * 8];
            c1[r] = s_pb[16 + r + lhalf * 8];
        }
        ABfrag bf;
        const _Float16* bp = &s_y[(nt * 16 + lmod) * 32 + koff];
        bf.h[0] = *(const v8h*)bp;
        bf.h[1] = *(const v8h*)(bp + 16);
        c0 = __builtin_amdgcn_wmma_f32_16x16x32_f16(
            false, a0.v, false, bf.v, (short)0, c0, false, false);
        c1 = __builtin_amdgcn_wmma_f32_16x16x32_f16(
            false, a1.v, false, bf.v, (short)0, c1, false, false);
        int px0 = nt * 16;
        int row = px0 >> 5;
        int colb = (px0 & 31) + lmod;
#pragma unroll
        for (int r = 0; r < 8; ++r) {
            int ch0 = r + lhalf * 8;
            int ch1 = 16 + ch0;
            size_t o0 = ((size_t)(b * 32 + ch0) * H + h0 + row) * W + w0 + colb;
            size_t o1 = ((size_t)(b * 32 + ch1) * H + h0 + row) * W + w0 + colb;
            out[o0] = c0[r];
            out[o1] = c1[r];
        }
    }
}

// ---------------------------------------------------------------------------
extern "C" void kernel_launch(void* const* d_in, const int* in_sizes, int n_in,
                              void* d_out, int out_size, void* d_ws, size_t ws_size,
                              hipStream_t stream) {
    (void)in_sizes; (void)n_in; (void)out_size; (void)ws_size;
    const float* x    = (const float*)d_in[0];
    const float* pa_w = (const float*)d_in[1];
    const float* pa_b = (const float*)d_in[2];
    const float* ex_w = (const float*)d_in[3];
    const float* ex_b = (const float*)d_in[4];
    const float* pw_w = (const float*)d_in[5];
    const float* pw_b = (const float*)d_in[6];
    float* out = (float*)d_out;

    char* ws = (char*)d_ws;
    float* featA = (float*)ws;                           // 8 MB max
    float* featB = (float*)(ws + ((size_t)8 << 20));     // 8 MB max
    float* wmap  = (float*)(ws + ((size_t)16 << 20));    // 21 MB

    avgpool4_k<<<(4 * 32 * 128 * 128) / 256, 256, 0, stream>>>(x, featA);

    for (int i = 0; i < 2; ++i) {                        // predictors @128x128
        conv3x3_wmma<<<dim3(64, 4), 256, 0, stream>>>(
            featA, nullptr, featB, pa_w + (size_t)(i * 2 + 0) * 9216,
            pa_b + (i * 2 + 0) * 32, 128, 128, 0);
        conv3x3_wmma<<<dim3(64, 4), 256, 0, stream>>>(
            featB, featA, featA, pa_w + (size_t)(i * 2 + 1) * 9216,
            pa_b + (i * 2 + 1) * 32, 128, 128, 1);
    }
    avgpool2_k<<<(4 * 32 * 64 * 64) / 256, 256, 0, stream>>>(featA, featB);
    for (int i = 2; i < 4; ++i) {                        // predictors @64x64
        conv3x3_wmma<<<dim3(16, 4), 256, 0, stream>>>(
            featB, nullptr, featA, pa_w + (size_t)(i * 2 + 0) * 9216,
            pa_b + (i * 2 + 0) * 32, 64, 64, 0);
        conv3x3_wmma<<<dim3(16, 4), 256, 0, stream>>>(
            featA, featB, featB, pa_w + (size_t)(i * 2 + 1) * 9216,
            pa_b + (i * 2 + 1) * 32, 64, 64, 1);
    }
    expand_tanh_k<<<(4 * 320 * 4096) / 256, 256, 0, stream>>>(featB, ex_w, ex_b, wmap);

    flying_pw_wmma<<<dim3(16, 64, 4), 256, FLY_LDS_BYTES, stream>>>(
        x, wmap, pw_w, pw_b, out);
}
